// ELKUNet_25417616458393
// MI455X (gfx1250) — compile-verified, matrix operand order checked by reading, not run on hardware
//
#include <hip/hip_runtime.h>

#define NPTS   200000
#define CCH    64
#define GDIM   256
#define GSD    32
#define NSEG   (GSD*GSD*GSD)
#define G3     (GDIM*GDIM*GDIM)
#define NTILES (NPTS/16)          // 12500 exactly (gemm_pre, 16-row tiles)
#define NT32   (NPTS/32)          // 6250 exactly  (conv, 32-row tiles)
#define LN_EPS 1e-6f

typedef _Float16 v16h __attribute__((ext_vector_type(16)));
typedef float    v8f  __attribute__((ext_vector_type(8)));

union V16H { v16h v; float4 f[2]; };

__device__ __forceinline__ float wsum32(float v) {
#pragma unroll
  for (int m = 16; m > 0; m >>= 1) v += __shfl_xor(v, m, 32);
  return v;
}

__device__ __forceinline__ v8f wmma16(V16H a, V16H b, v8f c) {
  return __builtin_amdgcn_wmma_f32_16x16x32_f16(false, a.v, false, b.v,
                                                (short)0, c, false, false);
}

// gather one 16-wide row pair of A fragments (K 0..31 and 32..63) for a neighbor
__device__ __forceinline__ void gatherA(const _Float16* __restrict__ fh,
                                        const int* __restrict__ grid,
                                        int nx, int ny, int nz, int klo,
                                        V16H* a0, V16H* a1) {
  bool inb = (unsigned)nx < GDIM && (unsigned)ny < GDIM && (unsigned)nz < GDIM;
  int nidx = inb ? grid[((size_t)nx * GDIM + ny) * GDIM + nz] : -1;
  bool valid = nidx >= 0;
  const float4* arow = (const float4*)(fh + (size_t)(valid ? nidx : 0) * CCH);
  float4 z4 = make_float4(0.f, 0.f, 0.f, 0.f);
  a0->f[0] = valid ? arow[(klo) >> 3]      : z4;
  a0->f[1] = valid ? arow[(klo + 16) >> 3] : z4;
  a1->f[0] = valid ? arow[(klo + 32) >> 3] : z4;
  a1->f[1] = valid ? arow[(klo + 48) >> 3] : z4;
}

// ---------------- prep kernels ----------------
__global__ void k_convert(const float* __restrict__ src, _Float16* __restrict__ dst, int n) {
  int i = blockIdx.x * blockDim.x + threadIdx.x;
  if (i < n) dst[i] = (_Float16)src[i];
}

// conv_w (27,C,C) row-major -> f16, transposed per tap: convT[k][j*C+i] = conv_w[k][i*C+j]
__global__ void k_convert_convw(const float* __restrict__ cw, _Float16* __restrict__ ct) {
  int i = blockIdx.x * blockDim.x + threadIdx.x;
  if (i < 27 * CCH * CCH) {
    int k = i >> 12, r = i & 4095;
    int ii = r >> 6, jj = r & 63;
    ct[(k << 12) + (jj << 6) + ii] = (_Float16)cw[i];
  }
}

__global__ void k_grid_init(int4* __restrict__ g) {
  int i = blockIdx.x * blockDim.x + threadIdx.x;
  if (i < G3 / 4) g[i] = make_int4(-1, -1, -1, -1);
}

__global__ void k_grid_scatter(const int* __restrict__ coords, int* __restrict__ grid) {
  int p = blockIdx.x * blockDim.x + threadIdx.x;
  if (p < NPTS) {
    int x = coords[p * 4], y = coords[p * 4 + 1], z = coords[p * 4 + 2];
    grid[((size_t)x * GDIM + y) * GDIM + z] = p;
  }
}

__global__ void k_zero(float* __restrict__ p, int n) {
  int i = blockIdx.x * blockDim.x + threadIdx.x;
  if (i < n) p[i] = 0.0f;
}

// ---------------- GEMM: F_pre = feats @ W_pre^T  (WMMA f16) ----------------
__global__ void k_gemm_pre(const _Float16* __restrict__ fh,
                           const _Float16* __restrict__ wh,
                           float* __restrict__ Fpre) {
  __shared__ __align__(16) _Float16 Bs[CCH * CCH];
  int tid = threadIdx.x;
  { // stage W_pre (B(K=i,N=j)=W_pre[j*64+i] -> row j contiguous in K already)
    const float4* src = (const float4*)wh;
    float4* dst = (float4*)Bs;
    for (int i = tid; i < CCH * CCH / 8; i += 256) dst[i] = src[i];
  }
  __syncthreads();
  int wave = tid >> 5, lane = tid & 31;
  int tile = blockIdx.x * 8 + wave;
  if (tile >= NTILES) return;

  int row = lane & 15;
  int klo = (lane & 16) ? 8 : 0;
  size_t p = (size_t)tile * 16 + row;
  const float4* arow = (const float4*)(fh + p * CCH);
  V16H a0, a1;
  a0.f[0] = arow[(klo) >> 3];      a0.f[1] = arow[(klo + 16) >> 3];
  a1.f[0] = arow[(klo + 32) >> 3]; a1.f[1] = arow[(klo + 48) >> 3];

  v8f zacc = {0.f,0.f,0.f,0.f,0.f,0.f,0.f,0.f};
  v8f acc[4];
#pragma unroll
  for (int nt = 0; nt < 4; nt++) {
    acc[nt] = zacc;
    int n = nt * 16 + (lane & 15);
    const float4* brow = (const float4*)(Bs + n * CCH);
    V16H b0, b1;
    b0.f[0] = brow[(klo) >> 3];      b0.f[1] = brow[(klo + 16) >> 3];
    b1.f[0] = brow[(klo + 32) >> 3]; b1.f[1] = brow[(klo + 48) >> 3];
    acc[nt] = wmma16(a0, b0, acc[nt]);
    acc[nt] = wmma16(a1, b1, acc[nt]);
  }
  int mbase = tile * 16 + ((lane & 16) ? 8 : 0);
#pragma unroll
  for (int nt = 0; nt < 4; nt++)
#pragma unroll
    for (int v = 0; v < 8; v++)
      Fpre[(size_t)(mbase + v) * CCH + nt * 16 + (lane & 15)] = acc[nt][v];
}

// ---------------- warp-per-row layernorm (in place) ----------------
__global__ void k_ln_rows(float* __restrict__ F, const float* __restrict__ w,
                          const float* __restrict__ b) {
  int wave = threadIdx.x >> 5, lane = threadIdx.x & 31;
  int row = blockIdx.x * 8 + wave;
  if (row >= NPTS) return;
  float* r = F + (size_t)row * CCH;
  float v0 = r[lane], v1 = r[lane + 32];
  float m = wsum32(v0 + v1) * (1.0f / 64.0f);
  float d0 = v0 - m, d1 = v1 - m;
  float var = wsum32(d0 * d0 + d1 * d1) * (1.0f / 64.0f);
  float rs = rsqrtf(var + LN_EPS);
  r[lane]      = d0 * rs * w[lane]      + b[lane];
  r[lane + 32] = d1 * rs * w[lane + 32] + b[lane + 32];
}

// ---------------- 27-tap gathered conv (WMMA f16, 32-row tiles, dbl-buffered B) ----------------
__global__ void k_conv(const _Float16* __restrict__ fh,
                       const _Float16* __restrict__ ct,
                       const int* __restrict__ coords,
                       const int* __restrict__ grid,
                       float* __restrict__ localb) {
  __shared__ __align__(16) _Float16 Bs[2][CCH * CCH];  // 2 x 8KB, double-buffered
  int tid = threadIdx.x, wave = tid >> 5, lane = tid & 31;
  int tile = blockIdx.x * 8 + wave;                    // 32-point tile
  bool active = tile < NT32;
  int rlo = lane & 15;
  int klo = (lane & 16) ? 8 : 0;
  int x0 = 0, y0 = 0, z0 = 0, x1 = 0, y1 = 0, z1 = 0;
  if (active) {
    int p0 = tile * 32 + rlo, p1 = p0 + 16;
    x0 = coords[p0 * 4]; y0 = coords[p0 * 4 + 1]; z0 = coords[p0 * 4 + 2];
    x1 = coords[p1 * 4]; y1 = coords[p1 * 4 + 1]; z1 = coords[p1 * 4 + 2];
  }
  v8f zacc = {0.f,0.f,0.f,0.f,0.f,0.f,0.f,0.f};
  v8f acc[8] = {zacc, zacc, zacc, zacc, zacc, zacc, zacc, zacc};

  { // stage tap 0
    const float4* src = (const float4*)ct;
    float4* dst = (float4*)Bs[0];
    dst[tid] = src[tid];
    dst[tid + 256] = src[tid + 256];
  }
  __syncthreads();

  for (int k = 0; k < 27; k++) {
    if (k + 1 < 27) {  // prefetch next tap into the other buffer (no sync needed yet)
      const float4* src = (const float4*)(ct + ((size_t)(k + 1) << 12));
      float4* dst = (float4*)Bs[(k + 1) & 1];
      dst[tid] = src[tid];
      dst[tid + 256] = src[tid + 256];
    }
    if (active) {
      int dx = k / 9 - 1, dy = (k / 3) % 3 - 1, dz = k % 3 - 1;
      V16H a0lo, a0hi, a1lo, a1hi;
      gatherA(fh, grid, x0 + dx, y0 + dy, z0 + dz, klo, &a0lo, &a0hi);
      gatherA(fh, grid, x1 + dx, y1 + dy, z1 + dz, klo, &a1lo, &a1hi);
      const _Float16* B = Bs[k & 1];
#pragma unroll
      for (int nt = 0; nt < 4; nt++) {
        int n = nt * 16 + rlo;
        const float4* brow = (const float4*)(B + n * CCH);
        V16H b0, b1;
        b0.f[0] = brow[(klo) >> 3];      b0.f[1] = brow[(klo + 16) >> 3];
        b1.f[0] = brow[(klo + 32) >> 3]; b1.f[1] = brow[(klo + 48) >> 3];
        acc[nt]     = wmma16(a0lo, b0, acc[nt]);
        acc[nt]     = wmma16(a0hi, b1, acc[nt]);
        acc[4 + nt] = wmma16(a1lo, b0, acc[4 + nt]);
        acc[4 + nt] = wmma16(a1hi, b1, acc[4 + nt]);
      }
    }
    __syncthreads();  // staging of k+1 done; compute on Bs[k&1] done
  }
  if (active) {
    int mb0 = tile * 32 + ((lane & 16) ? 8 : 0);
    int mb1 = mb0 + 16;
#pragma unroll
    for (int nt = 0; nt < 4; nt++)
#pragma unroll
      for (int v = 0; v < 8; v++) {
        localb[(size_t)(mb0 + v) * CCH + nt * 16 + rlo] = acc[nt][v];
        localb[(size_t)(mb1 + v) * CCH + nt * 16 + rlo] = acc[4 + nt][v];
      }
  }
}

// ---------------- segment-sum scatter (block of 192 = 3C per point) ----------------
__global__ void k_scatter(const float* __restrict__ F, const int* __restrict__ coords,
                          const float* __restrict__ Wpos, const float* __restrict__ alpha,
                          float* __restrict__ sums, float* __restrict__ counts) {
  int p = blockIdx.x;
  int t = threadIdx.x;               // 0..191
  int x = coords[p * 4], y = coords[p * 4 + 1], z = coords[p * 4 + 2];
  int seg = ((x >> 3) * GSD + (y >> 3)) * GSD + (z >> 3);
  int c = t & 63, grp = t >> 6;
  float pos = ((float)x * Wpos[c * 3] + (float)y * Wpos[c * 3 + 1] +
               (float)z * Wpos[c * 3 + 2]) * alpha[c];
  float Fi = F[(size_t)p * CCH + c];
  float val = (grp == 0) ? Fi * cosf(pos) : (grp == 1) ? Fi * sinf(pos) : Fi * pos;
  atomicAdd(&sums[(size_t)seg * 192 + t], val);
  if (t == 0) atomicAdd(&counts[seg], 1.0f);
}

// ---------------- fused final: vox mix + 2x layernorm + relu ----------------
__global__ void k_final(const float* __restrict__ F, const float* __restrict__ localb,
                        const int* __restrict__ coords,
                        const float* __restrict__ Wpos, const float* __restrict__ alpha,
                        const float* __restrict__ sums, const float* __restrict__ counts,
                        const float* __restrict__ lnw, const float* __restrict__ lnb,
                        const float* __restrict__ lnlw, const float* __restrict__ lnlb,
                        float* __restrict__ out) {
  int wave = threadIdx.x >> 5, lane = threadIdx.x & 31;
  int p = blockIdx.x * 8 + wave;
  if (p >= NPTS) return;
  int x = coords[p * 4], y = coords[p * 4 + 1], z = coords[p * 4 + 2];
  int seg = ((x >> 3) * GSD + (y >> 3)) * GSD + (z >> 3);
  float inv = 1.0f / fmaxf(counts[seg], 1.0f);
  const float* sg = &sums[(size_t)seg * 192];

  float nf[2], lc[2];
  int cs[2] = {lane, lane + 32};
#pragma unroll
  for (int j = 0; j < 2; j++) {
    int c = cs[j];
    float pos = ((float)x * Wpos[c * 3] + (float)y * Wpos[c * 3 + 1] +
                 (float)z * Wpos[c * 3 + 2]) * alpha[c];
    float pc = cosf(pos), ps = sinf(pos);
    float Fi = F[(size_t)p * CCH + c];
    float Fw = Fi * pos;
    float v0 = sg[c] * inv, v1 = sg[64 + c] * inv, v2 = sg[128 + c] * inv;
    nf[j] = v0 * pc + v1 * ps + (v2 - Fw);
    lc[j] = localb[(size_t)p * CCH + c];
  }
  // layernorm(new_F)
  float m1 = wsum32(nf[0] + nf[1]) * (1.0f / 64.0f);
  float d0 = nf[0] - m1, d1 = nf[1] - m1;
  float r1 = rsqrtf(wsum32(d0 * d0 + d1 * d1) * (1.0f / 64.0f) + LN_EPS);
  float a0 = d0 * r1 * lnw[lane]      + lnb[lane];
  float a1 = d1 * r1 * lnw[lane + 32] + lnb[lane + 32];
  // layernorm(local)
  float m2 = wsum32(lc[0] + lc[1]) * (1.0f / 64.0f);
  float e0 = lc[0] - m2, e1 = lc[1] - m2;
  float r2 = rsqrtf(wsum32(e0 * e0 + e1 * e1) * (1.0f / 64.0f) + LN_EPS);
  float b0 = e0 * r2 * lnlw[lane]      + lnlb[lane];
  float b1 = e1 * r2 * lnlw[lane + 32] + lnlb[lane + 32];

  out[(size_t)p * CCH + lane]      = fmaxf(a0 + b0, 0.0f);
  out[(size_t)p * CCH + lane + 32] = fmaxf(a1 + b1, 0.0f);
}

extern "C" void kernel_launch(void* const* d_in, const int* in_sizes, int n_in,
                              void* d_out, int out_size, void* d_ws, size_t ws_size,
                              hipStream_t stream) {
  (void)in_sizes; (void)n_in; (void)out_size; (void)ws_size;
  const float* feats  = (const float*)d_in[0];
  const float* W_pre  = (const float*)d_in[1];
  const float* lnp_w  = (const float*)d_in[2];
  const float* lnp_b  = (const float*)d_in[3];
  const float* W_pos  = (const float*)d_in[4];
  const float* alpha  = (const float*)d_in[5];
  const float* conv_w = (const float*)d_in[6];
  const float* ln_w   = (const float*)d_in[7];
  const float* ln_b   = (const float*)d_in[8];
  const float* lnl_w  = (const float*)d_in[9];
  const float* lnl_b  = (const float*)d_in[10];
  const int*   coords = (const int*)d_in[11];
  float* out = (float*)d_out;

  char* w = (char*)d_ws;
  auto carve = [&](size_t bytes) -> char* {
    char* p = w; w += (bytes + 255) & ~(size_t)255; return p;
  };
  int*      grid    = (int*)      carve((size_t)G3 * 4);          // 64 MB
  _Float16* feats_h = (_Float16*) carve((size_t)NPTS * CCH * 2);  // 25.6 MB
  _Float16* convT_h = (_Float16*) carve((size_t)27 * CCH * CCH * 2);
  _Float16* wpre_h  = (_Float16*) carve((size_t)CCH * CCH * 2);
  float*    Fin     = (float*)    carve((size_t)NPTS * CCH * 4);  // 51.2 MB
  float*    localb  = (float*)    carve((size_t)NPTS * CCH * 4);  // 51.2 MB
  float*    sums    = (float*)    carve((size_t)NSEG * 192 * 4);  // 25.2 MB
  float*    counts  = (float*)    carve((size_t)NSEG * 4);

  k_convert<<<(NPTS * CCH + 255) / 256, 256, 0, stream>>>(feats, feats_h, NPTS * CCH);
  k_convert<<<(CCH * CCH + 255) / 256, 256, 0, stream>>>(W_pre, wpre_h, CCH * CCH);
  k_convert_convw<<<(27 * CCH * CCH + 255) / 256, 256, 0, stream>>>(conv_w, convT_h);
  k_grid_init<<<(G3 / 4 + 255) / 256, 256, 0, stream>>>((int4*)grid);
  k_grid_scatter<<<(NPTS + 255) / 256, 256, 0, stream>>>(coords, grid);
  k_zero<<<(NSEG * 192 + 255) / 256, 256, 0, stream>>>(sums, NSEG * 192);
  k_zero<<<(NSEG + 255) / 256, 256, 0, stream>>>(counts, NSEG);

  k_gemm_pre<<<(NTILES + 7) / 8, 256, 0, stream>>>(feats_h, wpre_h, Fin);
  k_ln_rows<<<NPTS / 8, 256, 0, stream>>>(Fin, lnp_w, lnp_b);
  k_conv<<<(NT32 + 7) / 8, 256, 0, stream>>>(feats_h, convT_h, coords, grid, localb);
  k_scatter<<<NPTS, 192, 0, stream>>>(Fin, coords, W_pos, alpha, sums, counts);
  k_final<<<NPTS / 8, 256, 0, stream>>>(Fin, localb, coords, W_pos, alpha, sums, counts,
                                        ln_w, ln_b, lnl_w, lnl_b, out);
}
